// NHNmodel_88201448390762
// MI455X (gfx1250) — matrix operand order repaired and checked by reading, not compiled, not run on hardware
//
#include <hip/hip_runtime.h>
#include <hip/hip_bf16.h>

typedef __attribute__((ext_vector_type(16))) _Float16 v16h;
typedef __attribute__((ext_vector_type(8)))  float    v8f;

#define HID 64
#define NG  128

// ---------------------------------------------------------------------------
// Utility: zero a float buffer
__global__ void zero_kernel(float* p, int n) {
    int i = blockIdx.x * blockDim.x + threadIdx.x;
    if (i < n) p[i] = 0.f;
}

// ---------------------------------------------------------------------------
// Pre-swizzle W_ext = [w2 ; b2] (K_ext x 64) into f16 in the exact WMMA-B lane
// layout: element id = ((ch*4 + nt)*32 + lane)*16 + idx  corresponds to
// B[c][n] with c = ch*32 + (lane>=16?8:0) + ((idx>>3)<<4) + (idx&7), n = nt*16 + (lane&15).
__global__ void prep_wext_kernel(const float* __restrict__ w2,
                                 const float* __restrict__ b2,
                                 _Float16* __restrict__ wext,
                                 int IC, int chunks) {
    int id = blockIdx.x * blockDim.x + threadIdx.x;
    int total = chunks * 2048;
    if (id >= total) return;
    int idx  = id & 15;
    int lane = (id >> 4) & 31;
    int nt   = (id >> 9) & 3;
    int ch   = id >> 11;
    int c = ch * 32 + ((lane >= 16) ? 8 : 0) + ((idx >> 3) << 4) + (idx & 7);
    int n = nt * 16 + (lane & 15);
    float v = 0.f;
    int Kmain = 64 * IC;
    if (c < Kmain) {
        int k = c / IC, i = c % IC;
        v = w2[k * (IC * 64) + i * 64 + n];
    } else if (c < Kmain + IC) {
        v = b2[(c - Kmain) * 64 + n];
    }
    wext[id] = (_Float16)v;
}

// ---------------------------------------------------------------------------
// e = relu(edge_attr @ w1 + b1)   [E,7] @ [7,64]
__global__ void edge_mlp_kernel(const float* __restrict__ ea,
                                const float* __restrict__ w1,
                                const float* __restrict__ b1,
                                float* __restrict__ e_feat, int E) {
    int id = blockIdx.x * blockDim.x + threadIdx.x;
    if (id >= E * 64) return;
    int e = id >> 6, o = id & 63;
    float acc = b1[o];
#pragma unroll
    for (int j = 0; j < 7; ++j) acc += ea[e * 7 + j] * w1[j * 64 + o];
    e_feat[id] = fmaxf(acc, 0.f);
}

// ---------------------------------------------------------------------------
// Fused message GEMM:  msg = p_ext @ W_ext   (p built on the fly in registers)
// Block = 128 threads = 4 waves; each wave owns TWO 16-edge M-tiles (32 edges)
// so each loaded B fragment feeds 2 WMMAs -> halves L2 B-traffic per FLOP.
// Block covers 128 edges.
template<int IC>
__global__ __launch_bounds__(128)
void msg_wmma_kernel(const float* __restrict__ e_feat,   // [E,64]
                     const float* __restrict__ h,        // [N,IC]
                     const _Float16* __restrict__ wext,  // swizzled
                     const int* __restrict__ src,        // [E]
                     float* __restrict__ msg, int E) {
    constexpr int CHUNKS = (65 * IC + 31) / 32;   // 130 (IC=64), 3 (IC=1)
    __shared__ float e_s[128 * 64];
    __shared__ float h_s[128 * IC];

    int t = threadIdx.x;
    int wv = t >> 5, lane = t & 31;
    int blockBase = blockIdx.x * 128;

    // --- stage e tile [128 edges][64] ---
    if (blockBase + 128 <= E) {
        const float4* s4 = (const float4*)(e_feat + (size_t)blockBase * 64);
        float4* d4 = (float4*)e_s;
        for (int i = t; i < 128 * 16; i += 128) d4[i] = s4[i];
    } else {
        for (int i = t; i < 128 * 64; i += 128) {
            int eg = blockBase + (i >> 6);
            e_s[i] = (eg < E) ? e_feat[(size_t)eg * 64 + (i & 63)] : 0.f;
        }
    }
    // --- gather h rows for the 128 edges ---
    for (int i = t; i < 128 * IC; i += 128) {
        int el = i / IC, ii = i % IC;
        int eg = blockBase + el;
        h_s[i] = (eg < E) ? h[(size_t)src[eg] * IC + ii] : 0.f;
    }
    __syncthreads();

    int rbase = lane & 15;            // A-matrix M row within an M-tile
    int kb    = (lane >> 4) * 8;      // K sub-offset per A layout
    int waveEdge = blockBase + wv * 32;

    // Per-lane register cache of the h values each lane ever needs (IC=64):
    // 32 values per M-tile, 2 M-tiles.
    float hreg[2][32];
    if (IC == 64) {
#pragma unroll
        for (int mt = 0; mt < 2; ++mt) {
            int rloc = wv * 32 + mt * 16 + rbase;
#pragma unroll
            for (int par = 0; par < 2; ++par)
#pragma unroll
                for (int idx = 0; idx < 16; ++idx)
                    hreg[mt][par * 16 + idx] =
                        h_s[rloc * 64 + par * 32 + kb + ((idx >> 3) << 4) + (idx & 7)];
        }
    }

    v8f acc[2][4] = {};   // [M-tile][N-tile]

    for (int ch = 0; ch < CHUNKS; ++ch) {
        v16h a[2];
        if (IC == 64) {
            if (ch < 128) {               // main region: k constant per chunk
                int k = ch >> 1, par = ch & 1;
#pragma unroll
                for (int mt = 0; mt < 2; ++mt) {
                    float ev = e_s[(wv * 32 + mt * 16 + rbase) * 64 + k];
#pragma unroll
                    for (int idx = 0; idx < 16; ++idx)
                        a[mt][idx] = (_Float16)(ev * hreg[mt][par * 16 + idx]);
                }
            } else {                      // folded-b2 tail: p = h_src
                int par = ch & 1;
#pragma unroll
                for (int mt = 0; mt < 2; ++mt)
#pragma unroll
                    for (int idx = 0; idx < 16; ++idx)
                        a[mt][idx] = (_Float16)hreg[mt][par * 16 + idx];
            }
        } else {                          // IC==1: generic tiny path (3 chunks)
            int c0 = ch * 32;
#pragma unroll
            for (int mt = 0; mt < 2; ++mt) {
                int rloc = wv * 32 + mt * 16 + rbase;
#pragma unroll
                for (int idx = 0; idx < 16; ++idx) {
                    int c = c0 + kb + ((idx >> 3) << 4) + (idx & 7);
                    float v = 0.f;
                    if (c < 64 * IC)      v = e_s[rloc * 64 + c / IC] * h_s[rloc * IC + (c % IC)];
                    else if (c < 65 * IC) v = h_s[rloc * IC + (c - 64 * IC)];
                    a[mt][idx] = (_Float16)v;
                }
            }
        }
        // B operands: contiguous 32B per lane, pre-swizzled in global (L2-resident).
        // Each B fragment is reused for both M-tiles.
        const v16h* bp = (const v16h*)(wext + ((size_t)(ch * 4) * 32 + lane) * 16);
#pragma unroll
        for (int nt = 0; nt < 4; ++nt) {
            v16h b = bp[nt * 32];
#pragma unroll
            for (int mt = 0; mt < 2; ++mt)
                acc[mt][nt] = __builtin_amdgcn_wmma_f32_16x16x32_f16(
                    false, a[mt], false, b, (short)0, acc[mt][nt], false, false);
        }
    }

    // Store per documented 16x16 f32 C/D layout: lane<16 -> M=r, else M=r+8; N=lane&15
#pragma unroll
    for (int mt = 0; mt < 2; ++mt)
#pragma unroll
        for (int nt = 0; nt < 4; ++nt)
#pragma unroll
            for (int r = 0; r < 8; ++r) {
                int Mrow = (lane < 16) ? r : r + 8;
                int eg = waveEdge + mt * 16 + Mrow;
                if (eg < E)
                    msg[(size_t)eg * 64 + nt * 16 + (lane & 15)] = acc[mt][nt][r];
            }
}

// ---------------------------------------------------------------------------
// agg[dst] += msg  (segment sum via f32 global atomics)
__global__ void scatter_add_kernel(const float* __restrict__ msg,
                                   const int* __restrict__ dst,
                                   float* __restrict__ agg, int E) {
    int id = blockIdx.x * blockDim.x + threadIdx.x;
    if (id >= E * 64) return;
    int e = id >> 6;
    atomicAdd(&agg[(size_t)dst[e] * 64 + (id & 63)], msg[id]);
}

// h_out = relu(agg + h @ root + bias)
__global__ void node_update_kernel(const float* __restrict__ agg,
                                   const float* __restrict__ h,
                                   const float* __restrict__ root,
                                   const float* __restrict__ bias,
                                   float* __restrict__ hout, int N, int IC) {
    int id = blockIdx.x * blockDim.x + threadIdx.x;
    if (id >= N * 64) return;
    int n = id >> 6, o = id & 63;
    float acc = agg[id] + bias[o];
    for (int i = 0; i < IC; ++i) acc += h[(size_t)n * IC + i] * root[i * 64 + o];
    hout[id] = acc > 0.f ? acc : 0.f;
}

// ---------------------------------------------------------------------------
// Pooling: sum + max (h >= 0 post-ReLU so int-bit atomicMax is order-correct)
__global__ void pool_kernel(const float* __restrict__ h,
                            const int* __restrict__ batch,
                            float* __restrict__ sum, float* __restrict__ maxp,
                            int N) {
    int id = blockIdx.x * blockDim.x + threadIdx.x;
    if (id >= N * 64) return;
    int n = id >> 6, o = id & 63;
    float v = h[id];
    int g = batch[n];
    atomicAdd(&sum[g * 64 + o], v);
    atomicMax((int*)&maxp[g * 64 + o], __float_as_int(v));
}

__global__ void count_kernel(const int* __restrict__ batch, float* __restrict__ cnt, int N) {
    int n = blockIdx.x * blockDim.x + threadIdx.x;
    if (n < N) atomicAdd(&cnt[batch[n]], 1.f);
}

// g = concat(mean, max, sum)  -> [NG, 192]
__global__ void gbuild_kernel(const float* __restrict__ sum,
                              const float* __restrict__ maxp,
                              const float* __restrict__ cnt,
                              float* __restrict__ g) {
    int id = blockIdx.x * blockDim.x + threadIdx.x;
    if (id >= NG * 192) return;
    int r = id / 192, j = id % 192;
    float v;
    if (j < 64)       v = sum[r * 64 + j] / fmaxf(cnt[r], 1.f);
    else if (j < 128) v = maxp[r * 64 + (j - 64)];
    else              v = sum[r * 64 + (j - 128)];
    g[id] = v;
}

// ---------------------------------------------------------------------------
// Classifier layer: relu(layernorm(g @ w + b))   (block = one graph row)
__global__ void cls_layer_kernel(const float* __restrict__ gin,
                                 const float* __restrict__ w,
                                 const float* __restrict__ b,
                                 const float* __restrict__ lng,
                                 const float* __restrict__ lnb,
                                 float* __restrict__ gout, int P, int D) {
    __shared__ float gsrc[384];
    __shared__ float yrow[384];
    __shared__ float red[128];
    int r = blockIdx.x, t = threadIdx.x;
    for (int p = t; p < P; p += blockDim.x) gsrc[p] = gin[r * P + p];
    __syncthreads();
    for (int d = t; d < D; d += blockDim.x) {
        float acc = b[d];
        for (int p = 0; p < P; ++p) acc += gsrc[p] * w[p * D + d];
        yrow[d] = acc;
    }
    __syncthreads();
    float s = 0.f, sq = 0.f;
    for (int d = t; d < D; d += blockDim.x) { float v = yrow[d]; s += v; sq += v * v; }
    red[t] = s; __syncthreads();
    for (int off = 64; off; off >>= 1) { if (t < off) red[t] += red[t + off]; __syncthreads(); }
    float mu = red[0] / D; __syncthreads();
    red[t] = sq; __syncthreads();
    for (int off = 64; off; off >>= 1) { if (t < off) red[t] += red[t + off]; __syncthreads(); }
    float var = red[0] / D - mu * mu;
    float rstd = rsqrtf(var + 1e-5f);
    for (int d = t; d < D; d += blockDim.x) {
        float v = (yrow[d] - mu) * rstd * lng[d] + lnb[d];
        gout[r * D + d] = v > 0.f ? v : 0.f;
    }
}

__global__ void out_kernel(const float* __restrict__ g,
                           const float* __restrict__ w,
                           const float* __restrict__ b,
                           float* __restrict__ out) {
    int id = blockIdx.x * blockDim.x + threadIdx.x;
    if (id >= NG * 6) return;
    int r = id / 6, c = id % 6;
    float acc = b[c];
    for (int p = 0; p < 128; ++p) acc += g[r * 128 + p] * w[p * 6 + c];
    out[id] = acc;
}

// ---------------------------------------------------------------------------
extern "C" void kernel_launch(void* const* d_in, const int* in_sizes, int n_in,
                              void* d_out, int out_size, void* d_ws, size_t ws_size,
                              hipStream_t stream) {
    const float* x     = (const float*)d_in[0];     // [N,1]
    const int*   ei    = (const int*)d_in[1];       // [2,E]
    const float* ea    = (const float*)d_in[2];     // [E,7]
    const int*   batch = (const int*)d_in[3];       // [N]

    const int E = in_sizes[1] / 2;
    const int N = in_sizes[0];

    // workspace carve
    char* p = (char*)d_ws;
    auto alloc = [&](size_t bytes) -> void* {
        void* r = (void*)p;
        p += (bytes + 255) & ~(size_t)255;
        return r;
    };
    float*     h_a    = (float*)alloc((size_t)N * 64 * 4);
    float*     h_b    = (float*)alloc((size_t)N * 64 * 4);
    float*     e_feat = (float*)alloc((size_t)E * 64 * 4);
    float*     msg    = (float*)alloc((size_t)E * 64 * 4);
    float*     agg    = (float*)alloc((size_t)N * 64 * 4);
    _Float16*  wext   = (_Float16*)alloc((size_t)130 * 2048 * 2);
    float*     sump   = (float*)alloc((size_t)NG * 64 * 4);
    float*     maxp   = (float*)alloc((size_t)NG * 64 * 4);
    float*     cnt    = (float*)alloc((size_t)NG * 4);
    float*     g0     = (float*)alloc((size_t)NG * 384 * 4);
    float*     g1     = (float*)alloc((size_t)NG * 384 * 4);

    const int TB = 256;
    const float* h_in = x;
    float* h_out = h_a;

    for (int l = 0; l < 4; ++l) {
        const float* w1   = (const float*)d_in[4 + 6 * l + 0];
        const float* b1   = (const float*)d_in[4 + 6 * l + 1];
        const float* w2   = (const float*)d_in[4 + 6 * l + 2];
        const float* b2   = (const float*)d_in[4 + 6 * l + 3];
        const float* root = (const float*)d_in[4 + 6 * l + 4];
        const float* bias = (const float*)d_in[4 + 6 * l + 5];
        const int IC = (l == 0) ? 1 : 64;
        const int chunks = (65 * IC + 31) / 32;

        prep_wext_kernel<<<(chunks * 2048 + TB - 1) / TB, TB, 0, stream>>>(
            w2, b2, wext, IC, chunks);
        edge_mlp_kernel<<<((size_t)E * 64 + TB - 1) / TB, TB, 0, stream>>>(
            ea, w1, b1, e_feat, E);
        int mblocks = (E + 127) / 128;
        if (IC == 1)
            msg_wmma_kernel<1><<<mblocks, 128, 0, stream>>>(e_feat, h_in, wext, ei, msg, E);
        else
            msg_wmma_kernel<64><<<mblocks, 128, 0, stream>>>(e_feat, h_in, wext, ei, msg, E);
        zero_kernel<<<((size_t)N * 64 + TB - 1) / TB, TB, 0, stream>>>(agg, N * 64);
        scatter_add_kernel<<<((size_t)E * 64 + TB - 1) / TB, TB, 0, stream>>>(
            msg, ei + E, agg, E);
        node_update_kernel<<<((size_t)N * 64 + TB - 1) / TB, TB, 0, stream>>>(
            agg, h_in, root, bias, h_out, N, IC);
        h_in = h_out;
        h_out = (h_out == h_a) ? h_b : h_a;
    }

    // pooling
    zero_kernel<<<(NG * 64 + TB - 1) / TB, TB, 0, stream>>>(sump, NG * 64);
    zero_kernel<<<(NG * 64 + TB - 1) / TB, TB, 0, stream>>>(maxp, NG * 64);
    zero_kernel<<<(NG + TB - 1) / TB, TB, 0, stream>>>(cnt, NG);
    pool_kernel<<<((size_t)N * 64 + TB - 1) / TB, TB, 0, stream>>>(h_in, batch, sump, maxp, N);
    count_kernel<<<(N + TB - 1) / TB, TB, 0, stream>>>(batch, cnt, N);
    gbuild_kernel<<<(NG * 192 + TB - 1) / TB, TB, 0, stream>>>(sump, maxp, cnt, g0);

    // classifier
    const float* cw0 = (const float*)d_in[28]; const float* cb0 = (const float*)d_in[29];
    const float* cg0 = (const float*)d_in[30]; const float* cbb0 = (const float*)d_in[31];
    const float* cw1 = (const float*)d_in[32]; const float* cb1 = (const float*)d_in[33];
    const float* cg1 = (const float*)d_in[34]; const float* cbb1 = (const float*)d_in[35];
    const float* cw2 = (const float*)d_in[36]; const float* cb2 = (const float*)d_in[37];
    const float* cg2 = (const float*)d_in[38]; const float* cbb2 = (const float*)d_in[39];
    const float* ow  = (const float*)d_in[40]; const float* ob  = (const float*)d_in[41];

    cls_layer_kernel<<<NG, 128, 0, stream>>>(g0, cw0, cb0, cg0, cbb0, g1, 192, 384);
    cls_layer_kernel<<<NG, 128, 0, stream>>>(g1, cw1, cb1, cg1, cbb1, g0, 384, 256);
    cls_layer_kernel<<<NG, 128, 0, stream>>>(g0, cw2, cb2, cg2, cbb2, g1, 256, 128);
    out_kernel<<<(NG * 6 + TB - 1) / TB, TB, 0, stream>>>(g1, ow, ob, (float*)d_out);
}
